// NonLocalBlock_60172491817366
// MI455X (gfx1250) — matrix operand order repaired and checked by reading, not compiled, not run on hardware
//
#include <hip/hip_runtime.h>

#define BATCH 4
#define C_IN  256
#define ICH   128
#define NSP   4096
#define XSTR  48   // LDS row stride (elems) for x-tile: 96B, multiple of 32B

typedef __attribute__((ext_vector_type(16))) __bf16 v16bf;
typedef __attribute__((ext_vector_type(8)))  __bf16 v8bf;
typedef __attribute__((ext_vector_type(4)))  __bf16 v4bf;
typedef __attribute__((ext_vector_type(8)))  float  v8f;
typedef __attribute__((ext_vector_type(4)))  int    v4i;

static __device__ inline __bf16 f2bf(float f) { return (__bf16)f; }

static __device__ inline v8f wmma_bf16(v16bf a, v16bf b, v8f c) {
    // D = A(16x32 bf16) * B(32x16 bf16) + C(16x16 f32)
    return __builtin_amdgcn_wmma_f32_16x16x32_bf16(
        false, a, false, b, (short)0, c, false, false);
}

// A fragment: lane holds row m=l&15; K = {hb..hb+15} contiguous.
// p must point at row_base + hb (bf16 elems).
static __device__ inline v16bf ldA_bf(const __bf16* p) {
    v8bf lo = *(const v8bf*)p;
    v8bf hi = *(const v8bf*)(p + 8);
    v16bf r;
#pragma unroll
    for (int e = 0; e < 8; e++) { r[e] = lo[e]; r[e + 8] = hi[e]; }
    return r;
}

// ---- async global -> LDS copy (16B per lane), ASYNCcnt-tracked on gfx1250 ----
#if defined(__AMDGCN__) && __has_builtin(__builtin_amdgcn_global_load_async_to_lds_b128)
#define USE_ASYNC 1
#else
#define USE_ASYNC 0
#endif

static __device__ inline void cp16_g2l(__bf16* l, const __bf16* g) {
#if USE_ASYNC
    __builtin_amdgcn_global_load_async_to_lds_b128(
        (__attribute__((address_space(1))) v4i*)g,
        (__attribute__((address_space(3))) v4i*)l, 0, 0);
#else
    *(v8bf*)l = *(const v8bf*)g;   // VGPR-bounce fallback
#endif
}

#if USE_ASYNC
#if __has_builtin(__builtin_amdgcn_s_wait_asynccnt)
#define ASYNC_WAIT(n) __builtin_amdgcn_s_wait_asynccnt(n)
#else
#define ASYNC_WAIT(n) asm volatile("s_wait_asynccnt %0" :: "i"(n) : "memory")
#endif
#else
#define ASYNC_WAIT(n) ((void)0)
#endif

// -------------------------------------------------------------------------
// Prep: f32 -> bf16 bulk convert (memory-bound, ~1us total at 23.3 TB/s).
// -------------------------------------------------------------------------
__global__ void __launch_bounds__(256)
cvt_kernel(const float* __restrict__ src, __bf16* __restrict__ dst, int n4) {
    const int i = blockIdx.x * 256 + threadIdx.x;
    if (i < n4) {
        const float4 v = *(const float4*)(src + (size_t)i * 4);
        v4bf o;
        o[0] = f2bf(v.x); o[1] = f2bf(v.y); o[2] = f2bf(v.z); o[3] = f2bf(v.w);
        *(v4bf*)(dst + (size_t)i * 4) = o;
    }
}

// -------------------------------------------------------------------------
// Stage 1: theta/phi/g projections (1x1 convs), bf16 WMMA, K = C_IN = 256.
// Branch-free: every wave does 4 theta + 4 phi + 4 g tiles.
// Outputs: thetaT/phiT [B][N][IC], g [B][IC][N].
// -------------------------------------------------------------------------
__global__ void __launch_bounds__(256)
proj_kernel(const __bf16* __restrict__ xbf,
            const __bf16* __restrict__ thw, const float* __restrict__ thb,
            const __bf16* __restrict__ phw, const float* __restrict__ phb,
            const __bf16* __restrict__ gw,  const float* __restrict__ gb,
            __bf16* __restrict__ thetaT, __bf16* __restrict__ phiT,
            __bf16* __restrict__ gmat) {
    __shared__ __align__(128) __bf16 xT[64 * XSTR];   // x-tile transposed [i][c]
    const int b   = blockIdx.y;
    const int i0  = blockIdx.x * 64;
    const int tid = threadIdx.x;
    const int w   = tid >> 5;
    const int l   = tid & 31;
    const int n16 = l & 15;
    const int hb  = (l >> 4) * 16;
    const __bf16* xb = xbf + (size_t)b * C_IN * NSP;

    v8f acc[12];   // [0..3] theta, [4..7] phi, [8..11] g
#pragma unroll
    for (int t = 0; t < 12; t++)
#pragma unroll
        for (int e = 0; e < 8; e++) acc[t][e] = 0.f;

    for (int kc = 0; kc < 8; kc++) {
        const int c0 = kc * 32;
        __syncthreads();   // protect LDS reuse (prev iteration reads)
        // Cooperative load of xbf[c0..c0+32)[i0..i0+64) -> xT[i][c] (transpose)
        {
            const int row  = tid >> 3;            // c  0..31
            const int col8 = (tid & 7) * 8;       // i  0..56
            const v8bf v = *(const v8bf*)(xb + (size_t)(c0 + row) * NSP + i0 + col8);
#pragma unroll
            for (int e = 0; e < 8; e++)
                xT[(col8 + e) * XSTR + row] = v[e];
        }
        __syncthreads();

#pragma unroll
        for (int q = 0; q < 4; q++) {
            const int r   = w * 4 + q;             // 0..31, uniform per wave
            const int itA = r >> 3, icA = r & 7;   // theta/phi tile coords
            const int icG = r >> 2, itG = r & 3;   // g tile coords
            // theta: A = x^T rows (LDS), B = theta_w rows (global bf16)
            v16bf axt = ldA_bf(&xT[(itA * 16 + n16) * XSTR + hb]);
            const __bf16* tp = thw + (size_t)(icA * 16 + n16) * C_IN + c0 + (l >> 4) * 16;
            acc[q] = wmma_bf16(axt, *(const v16bf*)tp, acc[q]);
            // phi
            const __bf16* pp = phw + (size_t)(icA * 16 + n16) * C_IN + c0 + (l >> 4) * 16;
            acc[4 + q] = wmma_bf16(axt, *(const v16bf*)pp, acc[4 + q]);
            // g: A = g_w rows (global bf16), B = x^T columns (LDS)
            v16bf ag = ldA_bf(gw + (size_t)(icG * 16 + n16) * C_IN + c0 + hb);
            v16bf bx = *(const v16bf*)&xT[(itG * 16 + n16) * XSTR + (l >> 4) * 16];
            acc[8 + q] = wmma_bf16(ag, bx, acc[8 + q]);
        }
    }

    // Epilogue: bias + bf16 store in layouts matching later fragment loads.
#pragma unroll
    for (int q = 0; q < 4; q++) {
        const int r   = w * 4 + q;
        const int itA = r >> 3, icA = r & 7;
        const int icG = r >> 2, itG = r & 3;
        {   // theta
            const float bv = thb[icA * 16 + n16];
            __bf16* dst = thetaT + ((size_t)b * NSP + i0 + itA * 16) * ICH + icA * 16;
#pragma unroll
            for (int rr = 0; rr < 8; rr++)
                dst[(size_t)(rr + (l >> 4) * 8) * ICH + n16] = f2bf(acc[q][rr] + bv);
        }
        {   // phi
            const float bv = phb[icA * 16 + n16];
            __bf16* dst = phiT + ((size_t)b * NSP + i0 + itA * 16) * ICH + icA * 16;
#pragma unroll
            for (int rr = 0; rr < 8; rr++)
                dst[(size_t)(rr + (l >> 4) * 8) * ICH + n16] = f2bf(acc[4 + q][rr] + bv);
        }
        {   // g
            __bf16* dst = gmat + (size_t)b * ICH * NSP;
#pragma unroll
            for (int rr = 0; rr < 8; rr++) {
                const int ic = icG * 16 + rr + (l >> 4) * 8;
                dst[(size_t)ic * NSP + i0 + itG * 16 + n16] = f2bf(acc[8 + q][rr] + gb[ic]);
            }
        }
    }
}

// -------------------------------------------------------------------------
// Stage 2: flash-attention. Block = 8 waves; wave owns 16 rows (i-tile 128).
// phi/g j-tiles double-buffered in LDS via async DMA: next tile's
// global_load_async_to_lds_b128 issued before computing the current one,
// then s_wait_asynccnt<=8 (in-order completion => current tile resident).
// attn matrix never touches HBM.
// -------------------------------------------------------------------------
__global__ void __launch_bounds__(256)
attn_kernel(const __bf16* __restrict__ thetaT, const __bf16* __restrict__ phiT,
            const __bf16* __restrict__ gmat,   __bf16* __restrict__ oattn) {
    __shared__ __align__(128) __bf16 PH[2][64][ICH];   // phi tiles [j][ic] 32KB
    __shared__ __align__(128) __bf16 GM[2][ICH][64];   // g tiles   [ic][j] 32KB
    __shared__ __align__(128) __bf16 P[8][16 * 64];    // wave-private P    16KB
    const int b   = blockIdx.y;
    const int i0  = blockIdx.x * 128;
    const int tid = threadIdx.x;
    const int w   = tid >> 5;
    const int l   = tid & 31;
    const int n16 = l & 15;
    const int hb  = (l >> 4) * 16;

    // Stage one 64-j tile pair (phi + g) into buffer `buf` (8 async/wave).
    auto stage = [&](int buf, int j0) {
#pragma unroll
        for (int q = 0; q < 4; q++) {
            const int ch = q * 256 + tid;          // 0..1023
            cp16_g2l(&PH[buf][ch >> 4][(ch & 15) * 8],
                     phiT + ((size_t)b * NSP + j0 + (ch >> 4)) * ICH + (ch & 15) * 8);
        }
#pragma unroll
        for (int q = 0; q < 4; q++) {
            const int ch = q * 256 + tid;          // 0..1023
            cp16_g2l(&GM[buf][ch >> 3][(ch & 7) * 8],
                     gmat + ((size_t)b * ICH + (ch >> 3)) * NSP + j0 + (ch & 7) * 8);
        }
    };

    // Q fragments resident in registers: rows i = i0 + w*16 + (l&15), K=ic.
    v16bf qa[4];
#pragma unroll
    for (int kc = 0; kc < 4; kc++)
        qa[kc] = ldA_bf(thetaT + ((size_t)b * NSP + i0 + w * 16 + n16) * ICH + kc * 32 + hb);

    v8f o[8];
    float mrow[8], lrow[8];
#pragma unroll
    for (int ict = 0; ict < 8; ict++)
#pragma unroll
        for (int e = 0; e < 8; e++) o[ict][e] = 0.f;
#pragma unroll
    for (int rr = 0; rr < 8; rr++) { mrow[rr] = -1e30f; lrow[rr] = 0.f; }

    stage(0, 0);   // prologue

    for (int j0 = 0; j0 < NSP; j0 += 64) {
        const int cur = (j0 >> 6) & 1;
        if (j0 + 64 < NSP) {
            stage(cur ^ 1, j0 + 64);   // prefetch next tile (async, no wait)
            ASYNC_WAIT(8);             // this wave's current-tile loads done
        } else {
            ASYNC_WAIT(0);
        }
        __syncthreads();               // all waves' current-tile data visible

        // ---- S = Q . K^T : [16 x 64] logits in 4 D-fragments ----
        v8f s[4];
#pragma unroll
        for (int jt = 0; jt < 4; jt++) {
            v8f a;
#pragma unroll
            for (int e = 0; e < 8; e++) a[e] = 0.f;
#pragma unroll
            for (int kc = 0; kc < 4; kc++) {
                const __bf16* pb = &PH[cur][jt * 16 + n16][kc * 32 + (l >> 4) * 16];
                a = wmma_bf16(qa[kc], *(const v16bf*)pb, a);
            }
            s[jt] = a;
        }
        // ---- online softmax (row m = rr + (l>>4)*8; reduce over 16 n-lanes) ----
        float corr[8];
#pragma unroll
        for (int rr = 0; rr < 8; rr++) {
            float mx = fmaxf(fmaxf(s[0][rr], s[1][rr]), fmaxf(s[2][rr], s[3][rr]));
#pragma unroll
            for (int off = 1; off < 16; off <<= 1) mx = fmaxf(mx, __shfl_xor(mx, off, 32));
            const float nm = fmaxf(mrow[rr], mx);
            corr[rr] = __expf(mrow[rr] - nm);
            mrow[rr] = nm;
        }
#pragma unroll
        for (int rr = 0; rr < 8; rr++) {
            float ps = 0.f;
#pragma unroll
            for (int jt = 0; jt < 4; jt++) {
                const float p = __expf(s[jt][rr] - mrow[rr]);
                s[jt][rr] = p;
                ps += p;
            }
#pragma unroll
            for (int off = 1; off < 16; off <<= 1) ps += __shfl_xor(ps, off, 32);
            lrow[rr] = lrow[rr] * corr[rr] + ps;
#pragma unroll
            for (int ict = 0; ict < 8; ict++) o[ict][rr] = o[ict][rr] * corr[rr];
        }
        // ---- D-layout P -> LDS -> A-layout (wave-private, DS in-order) ----
#pragma unroll
        for (int jt = 0; jt < 4; jt++)
#pragma unroll
            for (int rr = 0; rr < 8; rr++)
                P[w][(rr + (l >> 4) * 8) * 64 + jt * 16 + n16] = f2bf(s[jt][rr]);

        v16bf pa[2];
#pragma unroll
        for (int k2 = 0; k2 < 2; k2++)
            pa[k2] = ldA_bf(&P[w][n16 * 64 + k2 * 32 + hb]);

        // ---- O += P . G (G fragments from LDS) ----
#pragma unroll
        for (int ict = 0; ict < 8; ict++) {
#pragma unroll
            for (int k2 = 0; k2 < 2; k2++) {
                const __bf16* gp = &GM[cur][ict * 16 + n16][k2 * 32 + (l >> 4) * 16];
                o[ict] = wmma_bf16(pa[k2], *(const v16bf*)gp, o[ict]);
            }
        }
        __syncthreads();   // readers done: buffer may be restaged next iter
    }
    // ---- finalize: O /= l, store bf16 [N][IC] ----
#pragma unroll
    for (int ict = 0; ict < 8; ict++)
#pragma unroll
        for (int rr = 0; rr < 8; rr++) {
            const int i = i0 + w * 16 + rr + (l >> 4) * 8;
            oattn[((size_t)b * NSP + i) * ICH + ict * 16 + n16] =
                f2bf(o[ict][rr] / lrow[rr]);
        }
}

// -------------------------------------------------------------------------
// Stage 3: W.O + bias, fused BatchNorm + residual. oattn tile staged once
// into LDS (all 8 waves share it); coalesced f32 stores.
// -------------------------------------------------------------------------
__global__ void __launch_bounds__(256)
outconv_kernel(const __bf16* __restrict__ oattn, const __bf16* __restrict__ ww,
               const float* __restrict__ wb, const float* __restrict__ gamma,
               const float* __restrict__ beta, const float* __restrict__ mean,
               const float* __restrict__ var, const float* __restrict__ x,
               float* __restrict__ out) {
    __shared__ __align__(128) __bf16 OA[64][ICH];    // oattn tile [i][ic] 16KB
    __shared__ float inv_s[C_IN], base_s[C_IN];
    const int b   = blockIdx.y;
    const int i0  = blockIdx.x * 64;
    const int tid = threadIdx.x;
    const int w   = tid >> 5;
    const int l   = tid & 31;
    const int n16 = l & 15;
    const int hb  = (l >> 4) * 16;

    {   // per-channel BN precompute: y = dot*inv + base + x
        const float iv = gamma[tid] * rsqrtf(var[tid] + 1e-5f);
        inv_s[tid]  = iv;
        base_s[tid] = wb[tid] * iv + beta[tid] - mean[tid] * iv;
    }
    // stage oattn[i0..i0+64)[0..128) -> LDS
#pragma unroll
    for (int q = 0; q < 4; q++) {
        const int ch = q * 256 + tid;              // 0..1023
        const int ir = ch >> 4;                    // i-local
        const int cc = (ch & 15) * 8;              // ic
        cp16_g2l(&OA[ir][cc], oattn + ((size_t)b * NSP + i0 + ir) * ICH + cc);
    }
    ASYNC_WAIT(0);
    __syncthreads();

    v8f acc[8];
#pragma unroll
    for (int t = 0; t < 8; t++)
#pragma unroll
        for (int e = 0; e < 8; e++) acc[t][e] = 0.f;

    for (int kc = 0; kc < 4; kc++) {
        v16bf bfrag[4];
#pragma unroll
        for (int it = 0; it < 4; it++)
            bfrag[it] = *(const v16bf*)&OA[it * 16 + n16][kc * 32 + (l >> 4) * 16];
#pragma unroll
        for (int ct = 0; ct < 2; ct++) {
            const int co16 = (w * 2 + ct) * 16;
            v16bf a = ldA_bf(ww + (size_t)(co16 + n16) * ICH + kc * 32 + hb);
#pragma unroll
            for (int it = 0; it < 4; it++)
                acc[ct * 4 + it] = wmma_bf16(a, bfrag[it], acc[ct * 4 + it]);
        }
    }
#pragma unroll
    for (int ct = 0; ct < 2; ct++)
#pragma unroll
        for (int it = 0; it < 4; it++) {
            const v8f a = acc[ct * 4 + it];
#pragma unroll
            for (int rr = 0; rr < 8; rr++) {
                const int co = (w * 2 + ct) * 16 + rr + (l >> 4) * 8;
                const int i  = i0 + it * 16 + n16;
                const size_t off = ((size_t)b * C_IN + co) * NSP + i;
                out[off] = a[rr] * inv_s[co] + base_s[co] + x[off];
            }
        }
}

extern "C" void kernel_launch(void* const* d_in, const int* in_sizes, int n_in,
                              void* d_out, int out_size, void* d_ws, size_t ws_size,
                              hipStream_t stream) {
    (void)in_sizes; (void)n_in; (void)out_size; (void)ws_size;
    const float* x     = (const float*)d_in[0];
    const float* thw   = (const float*)d_in[1];
    const float* thb   = (const float*)d_in[2];
    const float* phw   = (const float*)d_in[3];
    const float* phb   = (const float*)d_in[4];
    const float* gw    = (const float*)d_in[5];
    const float* gb    = (const float*)d_in[6];
    const float* ww    = (const float*)d_in[7];
    const float* wb    = (const float*)d_in[8];
    const float* gamma = (const float*)d_in[9];
    const float* beta  = (const float*)d_in[10];
    const float* mean  = (const float*)d_in[11];
    const float* var   = (const float*)d_in[12];

    const size_t nBNI = (size_t)BATCH * NSP * ICH;   // per bf16 activation tensor
    const size_t nX   = (size_t)BATCH * C_IN * NSP;
    const size_t nW   = (size_t)ICH * C_IN;          // 32768 (all weight mats)
    __bf16* thetaT = (__bf16*)d_ws;
    __bf16* phiT   = thetaT + nBNI;
    __bf16* gmat   = phiT + nBNI;
    __bf16* oattn  = gmat + nBNI;
    __bf16* xbf    = oattn + nBNI;
    __bf16* thwb   = xbf + nX;
    __bf16* phwb   = thwb + nW;
    __bf16* gwb    = phwb + nW;
    __bf16* wwb    = gwb + nW;

    // Prep: bulk f32 -> bf16 conversions (one-time, memory-bound).
    cvt_kernel<<<(int)(nX / 4 / 256), 256, 0, stream>>>(x, xbf, (int)(nX / 4));
    cvt_kernel<<<(int)(nW / 4 / 256), 256, 0, stream>>>(thw, thwb, (int)(nW / 4));
    cvt_kernel<<<(int)(nW / 4 / 256), 256, 0, stream>>>(phw, phwb, (int)(nW / 4));
    cvt_kernel<<<(int)(nW / 4 / 256), 256, 0, stream>>>(gw, gwb, (int)(nW / 4));
    cvt_kernel<<<(int)(nW / 4 / 256), 256, 0, stream>>>(ww, wwb, (int)(nW / 4));

    proj_kernel<<<dim3(NSP / 64, BATCH), 256, 0, stream>>>(
        xbf, thwb, thb, phwb, phb, gwb, gb, thetaT, phiT, gmat);
    attn_kernel<<<dim3(NSP / 128, BATCH), 256, 0, stream>>>(
        thetaT, phiT, gmat, oattn);
    outconv_kernel<<<dim3(NSP / 64, BATCH), 256, 0, stream>>>(
        oattn, wwb, wb, gamma, beta, mean, var, x, (float*)d_out);
}
